// ChameleonAttention_47897475285464
// MI455X (gfx1250) — matrix operand order repaired and checked by reading, not compiled, not run on hardware
//
#include <hip/hip_runtime.h>
#include <hip/hip_bf16.h>

#define S_LEN 2048
#define HID   4096
#define NHEAD 32
#define HD    128
#define LN_EPS 1e-5f

typedef __attribute__((ext_vector_type(16))) __bf16 bf16x16;
typedef __attribute__((ext_vector_type(8)))  float  f32x8;

union Frag  { bf16x16 v; uint4 u[2]; };
union Pack8 { uint4 u; __bf16 e[8]; };

__device__ inline f32x8 wmma_bf16(const Frag& a, const Frag& b, f32x8 c) {
  // D = A(16x32) * B(32x16) + C, bf16 in / f32 out
  return __builtin_amdgcn_wmma_f32_16x16x32_bf16(false, a.v, false, b.v,
                                                 (short)0, c, false, false);
}

// Async copy 16 bytes/lane from global straight into LDS (ASYNCcnt tracked).
// LDS address operand = low 32 bits of the generic pointer (flat pointers to
// LDS are {aperture_hi, lds_offset_lo}; ISA uses addr[31:0] as the LDS addr).
__device__ inline void async_load_b128(void* lds_ptr, const void* gptr) {
  unsigned ldsoff = (unsigned)(size_t)lds_ptr;
  asm volatile("global_load_async_to_lds_b128 %0, %1, off"
               :
               : "v"(ldsoff), "v"(gptr)
               : "memory");
}

// ---------------------------------------------------------------------------
// fp32 -> bf16 elementwise conversion
// ---------------------------------------------------------------------------
__global__ __launch_bounds__(256) void cvt_kernel(const float* __restrict__ in,
                                                  __bf16* __restrict__ out, int n) {
  int i = (blockIdx.x * blockDim.x + threadIdx.x) * 4;
  if (i + 3 < n) {
    float4 f = *(const float4*)&in[i];
    out[i + 0] = (__bf16)f.x;
    out[i + 1] = (__bf16)f.y;
    out[i + 2] = (__bf16)f.z;
    out[i + 3] = (__bf16)f.w;
  }
}

// ---------------------------------------------------------------------------
// GEMM: C[M,N] = A[M,K] * B[K,N], bf16 inputs, f32 accum.
// Block tile 64(M) x 128(N), 4 waves; each wave does 32x64 -> 8 WMMAs/K-step.
// Double-buffered LDS: A via global_load_async_to_lds_b128 (ASYNCcnt), B via
// register transpose; tile t+1 fetch overlaps tile t WMMA compute.
// ---------------------------------------------------------------------------
__global__ __launch_bounds__(128) void gemm_bf16_kernel(const __bf16* __restrict__ A,
                                                        const __bf16* __restrict__ B,
                                                        float* __restrict__ Cf,
                                                        __bf16* __restrict__ Cb,
                                                        int M, int N, int K) {
  __shared__ __bf16 sA[2][64 * 32];    // [m][k] row-major
  __shared__ __bf16 sBt[2][128 * 32];  // [n][k] transposed

  const int tid  = threadIdx.x;
  const int lane = tid & 31;
  const int hi   = (lane >> 4) & 1;
  const int ln   = lane & 15;
  const int wave = tid >> 5;   // 0..3
  const int wm   = wave >> 1;  // 0..1 : wave M tile (32 rows)
  const int wn   = wave & 1;   // 0..1 : wave N tile (64 cols)
  const int bm   = blockIdx.y * 64;
  const int bn   = blockIdx.x * 128;

  // Per-thread staging coordinates
  const int am  = tid >> 2;           // A chunk row (+32 for second chunk)
  const int akp = (tid & 3) * 8;      // A chunk k offset

  f32x8 acc[2][4];
  for (int i = 0; i < 2; ++i)
    for (int j = 0; j < 4; ++j)
      for (int r = 0; r < 8; ++r) acc[i][j][r] = 0.0f;

  // ---- prologue: stage tile 0 into buffer 0 ----
  async_load_b128(&sA[0][am * 32 + akp], &A[(size_t)(bm + am) * K + akp]);
  async_load_b128(&sA[0][(am + 32) * 32 + akp], &A[(size_t)(bm + am + 32) * K + akp]);
  {
    uint4 br[4];
    for (int it = 0; it < 4; ++it) {
      int c = tid + it * 128;
      br[it] = *(const uint4*)&B[(size_t)(c >> 4) * N + bn + ((c & 15) * 8)];
    }
    for (int it = 0; it < 4; ++it) {
      int c  = tid + it * 128;
      int kk = c >> 4;
      int np = (c & 15) * 8;
      Pack8 p; p.u = br[it];
      for (int t = 0; t < 8; ++t) sBt[0][(np + t) * 32 + kk] = p.e[t];
    }
  }
  asm volatile("s_wait_asynccnt 0x0" ::: "memory");
  __syncthreads();

  for (int kb = 0; kb < K; kb += 32) {
    const int buf  = (kb >> 5) & 1;
    const int nbuf = buf ^ 1;
    const bool more = (kb + 32) < K;

    // ---- issue tile t+1 fetch (overlaps with compute below) ----
    uint4 br[4];
    if (more) {
      int nk = kb + 32;
      async_load_b128(&sA[nbuf][am * 32 + akp],
                      &A[(size_t)(bm + am) * K + nk + akp]);
      async_load_b128(&sA[nbuf][(am + 32) * 32 + akp],
                      &A[(size_t)(bm + am + 32) * K + nk + akp]);
      for (int it = 0; it < 4; ++it) {
        int c = tid + it * 128;
        br[it] = *(const uint4*)&B[(size_t)(nk + (c >> 4)) * N + bn + ((c & 15) * 8)];
      }
      if (kb + 64 < K) {
        __builtin_prefetch(&B[(size_t)(kb + 64 + (tid >> 4)) * N + bn], 0, 1);
      }
    }

    // ---- compute on tile t ----
    Frag af[2];
    for (int msub = 0; msub < 2; ++msub) {
      int m    = wm * 32 + msub * 16 + ln;
      int base = hi * 16;
      af[msub].u[0] = *(const uint4*)&sA[buf][m * 32 + base];
      af[msub].u[1] = *(const uint4*)&sA[buf][m * 32 + base + 8];
    }
    for (int sub = 0; sub < 4; ++sub) {
      Frag bfrag;
      int nn = wn * 64 + sub * 16 + ln;
      const uint4* p = (const uint4*)&sBt[buf][nn * 32 + hi * 16];
      bfrag.u[0] = p[0];
      bfrag.u[1] = p[1];
      acc[0][sub] = wmma_bf16(af[0], bfrag, acc[0][sub]);
      acc[1][sub] = wmma_bf16(af[1], bfrag, acc[1][sub]);
    }

    // ---- finish staging tile t+1, then swap ----
    if (more) {
      for (int it = 0; it < 4; ++it) {
        int c  = tid + it * 128;
        int kk = c >> 4;
        int np = (c & 15) * 8;
        Pack8 p; p.u = br[it];
        for (int t = 0; t < 8; ++t) sBt[nbuf][(np + t) * 32 + kk] = p.e[t];
      }
      asm volatile("s_wait_asynccnt 0x0" ::: "memory");
    }
    __syncthreads();
  }

  for (int msub = 0; msub < 2; ++msub) {
    for (int sub = 0; sub < 4; ++sub) {
      int n = bn + wn * 64 + sub * 16 + ln;
      for (int r = 0; r < 8; ++r) {
        int m = bm + wm * 32 + msub * 16 + r + hi * 8;
        if (Cb) Cb[(size_t)m * N + n] = (__bf16)acc[msub][sub][r];
        else    Cf[(size_t)m * N + n] = acc[msub][sub][r];
      }
    }
  }
}

// ---------------------------------------------------------------------------
// Per-head layernorm + RoPE. One wave per row of 128 (4 elems / lane).
// in/out: [S, NH, D] bf16. gamma/beta: [D] f32.
// ---------------------------------------------------------------------------
__global__ __launch_bounds__(256) void ln_rope_kernel(const __bf16* __restrict__ in,
                                                      __bf16* __restrict__ out,
                                                      const float* __restrict__ gamma,
                                                      const float* __restrict__ beta,
                                                      const int* __restrict__ pos_ids) {
  int row  = blockIdx.x * 8 + (threadIdx.x >> 5);  // [0, S*NH)
  int lane = threadIdx.x & 31;
  int s    = row >> 5;  // row / NHEAD (NHEAD == 32)
  float pos = (float)pos_ids[s];

  const __bf16* rp = in + (size_t)row * HD + lane * 4;
  float x[4];
  for (int t = 0; t < 4; ++t) x[t] = (float)rp[t];

  float sum = x[0] + x[1] + x[2] + x[3];
  for (int m = 1; m < 32; m <<= 1) sum += __shfl_xor(sum, m, 32);
  float mean = sum * (1.0f / HD);

  float d2 = 0.0f;
  for (int t = 0; t < 4; ++t) { float d = x[t] - mean; d2 += d * d; }
  for (int m = 1; m < 32; m <<= 1) d2 += __shfl_xor(d2, m, 32);
  float rstd = rsqrtf(d2 * (1.0f / HD) + LN_EPS);

  float xn[4];
  for (int t = 0; t < 4; ++t) {
    int d = lane * 4 + t;
    xn[t] = (x[t] - mean) * rstd * gamma[d] + beta[d];
  }
  // rotate_half partner lives in lane ^ 16 (d +/- 64)
  for (int t = 0; t < 4; ++t) {
    int d = lane * 4 + t;
    float part = __shfl_xor(xn[t], 16, 32);
    int fb = d & 63;
    float freq = __powf(10000.0f, -(2.0f * (float)fb) * (1.0f / HD));
    float ang = pos * freq;
    float c = __cosf(ang), sn = __sinf(ang);
    float rh = (d < 64) ? -part : part;
    out[(size_t)row * HD + d] = (__bf16)(xn[t] * c + rh * sn);
  }
}

// ---------------------------------------------------------------------------
// Causal flash attention. One wave per (head, 16-query tile); 32-key blocks.
// Q/K/V/O: [S, NH, D] bf16. O holds the pre-Wo context (bf16).
// ---------------------------------------------------------------------------
__global__ __launch_bounds__(128) void attn_kernel(const __bf16* __restrict__ Q,
                                                   const __bf16* __restrict__ Kg,
                                                   const __bf16* __restrict__ Vg,
                                                   __bf16* __restrict__ O) {
  __shared__ __bf16 vts[4][HD * 32];  // V^T tile per wave: [d][key]
  __shared__ __bf16 pls[4][16 * 32];  // P tile per wave:  [m][key]

  const int tid  = threadIdx.x;
  const int lane = tid & 31;
  const int hi   = (lane >> 4) & 1;
  const int ln   = lane & 15;
  const int wv   = tid >> 5;
  __bf16* vtl = vts[wv];
  __bf16* pll = pls[wv];

  int w     = blockIdx.x * 4 + wv;  // 0..4095
  int head  = w >> 7;               // 32 heads
  int qt    = w & 127;              // 128 query tiles
  int qbase = qt * 16;

  // Q fragments (16x128 = 4 x 16x32 A fragments), straight from global
  Frag qf[4];
  {
    size_t rb = ((size_t)(qbase + ln) * NHEAD + head) * HD;
    for (int c = 0; c < 4; ++c) {
      int off = c * 32 + hi * 16;
      qf[c].u[0] = *(const uint4*)&Q[rb + off];
      qf[c].u[1] = *(const uint4*)&Q[rb + off + 8];
    }
  }

  f32x8 oacc[8];
  for (int i = 0; i < 8; ++i)
    for (int r = 0; r < 8; ++r) oacc[i][r] = 0.0f;
  float mrow[8], lrow[8];
  for (int r = 0; r < 8; ++r) { mrow[r] = -3.0e38f; lrow[r] = 0.0f; }

  const float scale = 0.0883883476483184f;  // 1/sqrt(128)
  int nkb = (qbase + 47) >> 5;              // key blocks covering keys <= qbase+15

  for (int kb = 0; kb < nkb; ++kb) {
    int keybase = kb * 32;

    // Stage V^T (32 keys x 128 d -> [d][key]) into LDS
    for (int i = 0; i < 16; ++i) {
      int c   = i * 32 + lane;  // 512 chunks of 8
      int key = c >> 4;
      int dp  = (c & 15) * 8;
      Pack8 p;
      p.u = *(const uint4*)&Vg[((size_t)(keybase + key) * NHEAD + head) * HD + dp];
      for (int t = 0; t < 8; ++t) vtl[(dp + t) * 32 + key] = p.e[t];
    }
    asm volatile("s_wait_dscnt 0" ::: "memory");

    // S = Q * K^T : two 16-key subtiles, accumulate over 4 K-dim chunks
    f32x8 s0, s1;
    for (int r = 0; r < 8; ++r) { s0[r] = 0.0f; s1[r] = 0.0f; }
    for (int c = 0; c < 4; ++c) {
      Frag kf0, kf1;
      const uint4* p0 = (const uint4*)
          &Kg[((size_t)(keybase + ln) * NHEAD + head) * HD + c * 32 + hi * 16];
      kf0.u[0] = p0[0]; kf0.u[1] = p0[1];
      const uint4* p1 = (const uint4*)
          &Kg[((size_t)(keybase + 16 + ln) * NHEAD + head) * HD + c * 32 + hi * 16];
      kf1.u[0] = p1[0]; kf1.u[1] = p1[1];
      s0 = wmma_bf16(qf[c], kf0, s0);
      s1 = wmma_bf16(qf[c], kf1, s1);
    }

    // Mask + scale + online softmax (row = r + hi*8, col = ln within subtile)
    float p0v[8], p1v[8], corr[8];
    for (int r = 0; r < 8; ++r) {
      int qrow = qbase + r + hi * 8;
      int k0 = keybase + ln;
      int k1 = k0 + 16;
      float a = (k0 <= qrow) ? s0[r] * scale : -3.0e38f;
      float b = (k1 <= qrow) ? s1[r] * scale : -3.0e38f;
      float mx = fmaxf(a, b);
      for (int m = 1; m < 16; m <<= 1) mx = fmaxf(mx, __shfl_xor(mx, m, 32));
      float mn = fmaxf(mrow[r], mx);
      corr[r] = __expf(mrow[r] - mn);
      float e0 = __expf(a - mn);
      float e1 = __expf(b - mn);
      float sm = e0 + e1;
      for (int m = 1; m < 16; m <<= 1) sm += __shfl_xor(sm, m, 32);
      lrow[r] = lrow[r] * corr[r] + sm;
      mrow[r] = mn;
      p0v[r] = e0;
      p1v[r] = e1;
    }
    for (int i = 0; i < 8; ++i)
      for (int r = 0; r < 8; ++r) oacc[i][r] *= corr[r];

    // Stage P (16x32) as bf16, reload as A fragment
    for (int r = 0; r < 8; ++r) {
      int m = r + hi * 8;
      pll[m * 32 + ln]      = (__bf16)p0v[r];
      pll[m * 32 + 16 + ln] = (__bf16)p1v[r];
    }
    asm volatile("s_wait_dscnt 0" ::: "memory");
    Frag pf;
    {
      int base = hi * 16;
      pf.u[0] = *(const uint4*)&pll[ln * 32 + base];
      pf.u[1] = *(const uint4*)&pll[ln * 32 + base + 8];
    }
    // O += P * V : 8 d-subtiles of 16
    for (int sub = 0; sub < 8; ++sub) {
      Frag vf;
      int d = sub * 16 + ln;
      const uint4* p = (const uint4*)&vtl[d * 32 + hi * 16];
      vf.u[0] = p[0];
      vf.u[1] = p[1];
      oacc[sub] = wmma_bf16(pf, vf, oacc[sub]);
    }
  }

  // epilogue: normalize by row sums and store bf16 context
  for (int r = 0; r < 8; ++r) {
    float inv = 1.0f / lrow[r];
    size_t rb = ((size_t)(qbase + r + hi * 8) * NHEAD + head) * HD;
    for (int sub = 0; sub < 8; ++sub)
      O[rb + sub * 16 + ln] = (__bf16)(oacc[sub][r] * inv);
  }
}

// ---------------------------------------------------------------------------
extern "C" void kernel_launch(void* const* d_in, const int* in_sizes, int n_in,
                              void* d_out, int out_size, void* d_ws, size_t ws_size,
                              hipStream_t stream) {
  const float* Xf  = (const float*)d_in[0];
  const int*   pos = (const int*)  d_in[1];
  const float* Wq  = (const float*)d_in[2];
  const float* Wk  = (const float*)d_in[3];
  const float* Wv  = (const float*)d_in[4];
  const float* Wo  = (const float*)d_in[5];
  const float* qnw = (const float*)d_in[6];
  const float* qnb = (const float*)d_in[7];
  const float* knw = (const float*)d_in[8];
  const float* knb = (const float*)d_in[9];
  float* out = (float*)d_out;

  char* ws = (char*)d_ws;
  size_t off = 0;
  auto alloc = [&](size_t bytes) -> char* {
    char* p = ws + off;
    off += (bytes + 255) & ~(size_t)255;
    return p;
  };
  const size_t SX = (size_t)S_LEN * HID;  // activations
  const size_t SW = (size_t)HID * HID;    // weights

  __bf16* Xb   = (__bf16*)alloc(SX * 2);
  __bf16* Wqb  = (__bf16*)alloc(SW * 2);
  __bf16* Wkb  = (__bf16*)alloc(SW * 2);
  __bf16* Wvb  = (__bf16*)alloc(SW * 2);
  __bf16* Wob  = (__bf16*)alloc(SW * 2);
  __bf16* Qraw = (__bf16*)alloc(SX * 2);
  __bf16* Kraw = (__bf16*)alloc(SX * 2);
  __bf16* Vraw = (__bf16*)alloc(SX * 2);
  __bf16* Qr   = (__bf16*)alloc(SX * 2);
  __bf16* Kr   = (__bf16*)alloc(SX * 2);
  __bf16* Ab   = (__bf16*)alloc(SX * 2);

  const int thr = 256;
  cvt_kernel<<<dim3((unsigned)(SX / (thr * 4))), thr, 0, stream>>>(Xf, Xb, (int)SX);
  cvt_kernel<<<dim3((unsigned)(SW / (thr * 4))), thr, 0, stream>>>(Wq, Wqb, (int)SW);
  cvt_kernel<<<dim3((unsigned)(SW / (thr * 4))), thr, 0, stream>>>(Wk, Wkb, (int)SW);
  cvt_kernel<<<dim3((unsigned)(SW / (thr * 4))), thr, 0, stream>>>(Wv, Wvb, (int)SW);
  cvt_kernel<<<dim3((unsigned)(SW / (thr * 4))), thr, 0, stream>>>(Wo, Wob, (int)SW);

  dim3 gg(HID / 128, S_LEN / 64);
  gemm_bf16_kernel<<<gg, 128, 0, stream>>>(Xb, Wqb, nullptr, Qraw, S_LEN, HID, HID);
  gemm_bf16_kernel<<<gg, 128, 0, stream>>>(Xb, Wkb, nullptr, Kraw, S_LEN, HID, HID);
  gemm_bf16_kernel<<<gg, 128, 0, stream>>>(Xb, Wvb, nullptr, Vraw, S_LEN, HID, HID);

  dim3 lr((unsigned)((size_t)S_LEN * NHEAD / 8));
  ln_rope_kernel<<<lr, 256, 0, stream>>>(Qraw, Qr, qnw, qnb, pos);
  ln_rope_kernel<<<lr, 256, 0, stream>>>(Kraw, Kr, knw, knb, pos);

  attn_kernel<<<dim3((S_LEN / 16) * NHEAD / 4), 128, 0, stream>>>(Qr, Kr, Vraw, Ab);

  gemm_bf16_kernel<<<gg, 128, 0, stream>>>(Ab, Wob, out, nullptr, S_LEN, HID, HID);
}
